// Siren_skip_hash_74182675136865
// MI455X (gfx1250) — matrix operand order repaired and checked by reading, not compiled, not run on hardware
//
#include <hip/hip_runtime.h>
#include <hip/hip_bf16.h>

typedef __attribute__((ext_vector_type(16))) _Float16 v16h;
typedef __attribute__((ext_vector_type(8)))  _Float16 v8h;
typedef __attribute__((ext_vector_type(8)))  float    v8f;

// -------- weight workspace: fragment-contiguous f16 layout ------------------
// Per layer: 32 N-tiles x NKB K-blocks; each fragment = 32 lanes x 16 f16
// contiguous (512 f16).  Element e of lane L in fragment (tileN, kb) is
//   W[n][k],  n = tileN*16 + (L&15),  k = kb*32 + ((L>>4)<<4) + e   (0 if k>=K)
#define W1F_OFF   0                          // 32 tiles x  1 kb  (K 22->32)
#define W2F_OFF   (32 * 1 * 512)             // 32 tiles x 16 kb
#define WS1F_OFF  (W2F_OFF + 32 * 16 * 512)  // 32 tiles x 17 kb  (K 534->544)
#define WS2F_OFF  (WS1F_OFF + 32 * 17 * 512) // 32 tiles x 16 kb
// total = 819200 f16 = 1,638,400 bytes of d_ws

template <int NKB, int KREAL>
__device__ __forceinline__ void prep_frag(const float* __restrict__ W,
                                          _Float16* __restrict__ dst, int i)
{
    const int frag = i >> 9;           // /512
    const int r    = i & 511;
    const int lane = r >> 4, e = r & 15;
    const int tileN = frag / NKB, kb = frag - tileN * NKB;
    const int n = tileN * 16 + (lane & 15);
    const int k = kb * 32 + ((lane >> 4) << 4) + e;
    dst[i] = (_Float16)(k < KREAL ? W[n * KREAL + k] : 0.0f);
}

__global__ __launch_bounds__(256) void prep_weights_kernel(
    const float* __restrict__ w1, const float* __restrict__ w2,
    const float* __restrict__ ws1, const float* __restrict__ ws2,
    _Float16* __restrict__ wb)
{
    const int i = blockIdx.x * 256 + threadIdx.x;
    if (i < 32 * 1 * 512)  prep_frag<1, 22>(w1, wb + W1F_OFF, i);
    if (i < 32 * 16 * 512) {
        prep_frag<16, 512>(w2,  wb + W2F_OFF,  i);
        prep_frag<16, 512>(ws2, wb + WS2F_OFF, i);
    }
    if (i < 32 * 17 * 512) prep_frag<17, 534>(ws1, wb + WS1F_OFF, i);
}

__device__ __forceinline__ v8f wmma_f16(v16h a, v16h b, v8f c)
{
    return __builtin_amdgcn_wmma_f32_16x16x32_f16(false, a, false, b,
                                                  (short)0, c, false, false);
}

// A fragment 16x32 f16 from LDS: per lane two contiguous 16B runs
// (k = hh..hh+7 and hh+8..hh+15), loaded as two ds_load_b128.
__device__ __forceinline__ v16h load_a_frag(const _Float16* A, int ld,
                                            int r0, int k0, int lane)
{
    const int m  = r0 + (lane & 15);
    const int hh = (lane >> 4) * 16;
    const _Float16* p = A + m * ld + k0 + hh;
    v8h lo = *(const v8h*)(p);
    v8h hi = *(const v8h*)(p + 8);
    return __builtin_shufflevector(lo, hi, 0, 1, 2, 3, 4, 5, 6, 7,
                                   8, 9, 10, 11, 12, 13, 14, 15);
}

// One layer for one wave: FOUR 16-row stripes (rows 0..63), 4 output tiles.
// Per K-block: stage all 4 B fragments, then 16 WMMAs (4 stripes x 4 tiles).
// Each B fragment feeds 4 WMMAs; each A fragment feeds 4 WMMAs.
template <int NKB>
__device__ __forceinline__ void wave_layer(
    const _Float16* __restrict__ A, int ldA,
    _Float16* __restrict__ O, int ldO,
    const _Float16* __restrict__ Bf, const float* __restrict__ bias,
    int tile0, int lane)
{
    v8f acc[4][4] = {};   // [row stripe][col tile]
    const _Float16* bp = Bf + (size_t)tile0 * NKB * 512 + lane * 16;
#pragma unroll
    for (int kb = 0; kb < NKB; ++kb) {
        v16h b[4];
#pragma unroll
        for (int t = 0; t < 4; ++t)
            b[t] = *(const v16h*)(bp + ((size_t)t * NKB + kb) * 512);
#pragma unroll
        for (int s = 0; s < 4; ++s) {
            const v16h a = load_a_frag(A, ldA, s * 16, kb * 32, lane);
#pragma unroll
            for (int t = 0; t < 4; ++t)
                acc[s][t] = wmma_f16(a, b[t], acc[s][t]);
        }
    }
    const int nl = lane & 15;
    const int mo = (lane >> 4) * 8;
#pragma unroll
    for (int t = 0; t < 4; ++t) {
        const int n = (tile0 + t) * 16 + nl;
        const float bv = bias[n];
#pragma unroll
        for (int s = 0; s < 4; ++s) {
#pragma unroll
            for (int r = 0; r < 8; ++r)
                O[(s * 16 + mo + r) * ldO + n] =
                    (_Float16)__sinf(acc[s][t][r] + bv);
        }
    }
}

#define ROWS 64
#define BLDA 520   // bufA leading dim (>=512, mult of 8 for 16B-aligned rows)
#define BLDB 552   // bufB leading dim (>=544, mult of 8)

__global__ __launch_bounds__(256) void siren_skip_hash_kernel(
    const float* __restrict__ coords, const float* __restrict__ table,
    const float* __restrict__ b1,  const float* __restrict__ b2,
    const float* __restrict__ bs1, const float* __restrict__ bs2,
    const float* __restrict__ wo,  const float* __restrict__ bo,
    const _Float16* __restrict__ wb, float* __restrict__ out)
{
    // bufB cols 512..543 permanently hold h0 (L1 input AND skip-concat tail);
    // cols 0..511 hold layer outputs.  bufA holds 512-wide activations.
    // 64*(520+552)*2 = 137,216 B static LDS (gfx1250: up to 320KB/WG).
    __shared__ __align__(16) _Float16 bufA[ROWS * BLDA];
    __shared__ __align__(16) _Float16 bufB[ROWS * BLDB];

    const int tid  = threadIdx.x;
    const int lane = tid & 31;
    const int wave = tid >> 5;
    const int rowBase = blockIdx.x * ROWS;

    // ---------------- hash encoding into bufB[:, 512:544] ------------------
    {
        constexpr int RESA[10] = {16, 22, 31, 43, 60, 84, 117, 164, 229, 319};
        constexpr unsigned PY = 2654435761u;
        for (int idx = tid; idx < ROWS * 10; idx += 256) {
            const int r = idx / 10, l = idx % 10;
            const int gr = rowBase + r;
            const float cx = coords[gr * 2 + 0];
            const float cy = coords[gr * 2 + 1];
            const float res = (float)RESA[l];
            const float xs = cx * res, ys = cy * res;
            const float xf = floorf(xs), yf = floorf(ys);
            const float wx = xs - xf, wy = ys - yf;
            const unsigned xi = (unsigned)(int)xf;
            const unsigned yi = (unsigned)(int)yf;
            const unsigned h00 = (xi        ^ (yi        * PY)) & 4095u;
            const unsigned h10 = ((xi + 1u) ^ (yi        * PY)) & 4095u;
            const unsigned h01 = (xi        ^ ((yi + 1u) * PY)) & 4095u;
            const unsigned h11 = ((xi + 1u) ^ ((yi + 1u) * PY)) & 4095u;
            const float* tab = table + l * 8192;
            const float f00x = tab[h00 * 2], f00y = tab[h00 * 2 + 1];
            const float f10x = tab[h10 * 2], f10y = tab[h10 * 2 + 1];
            const float f01x = tab[h01 * 2], f01y = tab[h01 * 2 + 1];
            const float f11x = tab[h11 * 2], f11y = tab[h11 * 2 + 1];
            const float fx0x = f00x + (f10x - f00x) * wx;
            const float fx0y = f00y + (f10y - f00y) * wx;
            const float fx1x = f01x + (f11x - f01x) * wx;
            const float fx1y = f01y + (f11y - f01y) * wx;
            bufB[r * BLDB + 512 + 2 * l]     = (_Float16)(fx0x + (fx1x - fx0x) * wy);
            bufB[r * BLDB + 512 + 2 * l + 1] = (_Float16)(fx0y + (fx1y - fx0y) * wy);
        }
        // cols 532,533 = coords; 534..543 = 0 (K padding for both L1 and S1)
        for (int idx = tid; idx < ROWS * 12; idx += 256) {
            const int r = idx / 12, j = idx % 12;
            float v = 0.0f;
            if (j < 2) v = coords[(rowBase + r) * 2 + j];
            bufB[r * BLDB + 532 + j] = (_Float16)v;
        }
    }
    __syncthreads();

    const int tile0 = wave * 4;

    // layer 1: h0 (bufB+512, K=32) -> bufA
    wave_layer<1>(bufB + 512, BLDB, bufA, BLDA, wb + W1F_OFF, b1, tile0, lane);
    __syncthreads();
    // layer 2: bufA (K=512) -> bufB[:, 0:512]   (h0 tail stays intact)
    wave_layer<16>(bufA, BLDA, bufB, BLDB, wb + W2F_OFF, b2, tile0, lane);
    __syncthreads();
    // skip layer 1: bufB (K=544, concat already in place) -> bufA
    wave_layer<17>(bufB, BLDB, bufA, BLDA, wb + WS1F_OFF, bs1, tile0, lane);
    __syncthreads();
    // skip layer 2: bufA (K=512) -> bufB[:, 0:512]
    wave_layer<16>(bufA, BLDA, bufB, BLDB, wb + WS2F_OFF, bs2, tile0, lane);
    __syncthreads();

    // ---------------- output layer: 512 -> 2 (VALU dot + shuffles) ----------
    {
        const int task = tid >> 1;   // 128 tasks = 64 rows x 2 outputs
        const int l2   = tid & 1;
        const int r = task >> 1, o = task & 1;
        float sum = 0.0f;
        const int kbase = l2 * 256;
#pragma unroll 8
        for (int k = kbase; k < kbase + 256; ++k)
            sum += (float)bufB[r * BLDB + k] * wo[o * 512 + k];
        sum += __shfl_xor(sum, 1, 2);
        if (l2 == 0)
            out[(rowBase + r) * 2 + o] = __sinf(sum + bo[o]);
    }
}

extern "C" void kernel_launch(void* const* d_in, const int* in_sizes, int n_in,
                              void* d_out, int out_size, void* d_ws, size_t ws_size,
                              hipStream_t stream)
{
    const float* coords = (const float*)d_in[0];
    const float* table  = (const float*)d_in[1];
    const float* w1     = (const float*)d_in[2];
    const float* b1     = (const float*)d_in[3];
    const float* w2     = (const float*)d_in[4];
    const float* b2     = (const float*)d_in[5];
    const float* ws1    = (const float*)d_in[6];
    const float* bs1    = (const float*)d_in[7];
    const float* ws2    = (const float*)d_in[8];
    const float* bs2    = (const float*)d_in[9];
    const float* wo     = (const float*)d_in[10];
    const float* bo     = (const float*)d_in[11];

    _Float16* wb = (_Float16*)d_ws;      // needs 1,638,400 bytes
    float* out = (float*)d_out;

    // 1) swizzle + convert weights into fragment-contiguous f16 layout
    prep_weights_kernel<<<(32 * 17 * 512 + 255) / 256, 256, 0, stream>>>(
        w1, w2, ws1, ws2, wb);

    // 2) fused hash-encode + 5-layer SIREN forward; 64 rows per block
    const int N = 262144;
    siren_skip_hash_kernel<<<N / ROWS, 256, 0, stream>>>(
        coords, table, b1, b2, bs1, bs2, wo, bo, wb, out);
}
